// FSRAMATModel_16939351015839
// MI455X (gfx1250) — matrix operand order repaired and checked by reading, not compiled, not run on hardware
//
#include <hip/hip_runtime.h>
#include <hip/hip_bf16.h>

// ---------------------------------------------------------------------------
// Types for CDNA5 WMMA (wave32): v_wmma_f32_16x16x32_bf16
// ---------------------------------------------------------------------------
typedef __attribute__((ext_vector_type(16))) __bf16 bf16x16;
typedef __attribute__((ext_vector_type(8)))  __bf16 bf16x8;
typedef __attribute__((ext_vector_type(8)))  float  f32x8;

// ---------------------------------------------------------------------------
// WMMA fragment layouts per CDNA5 ISA (CDNA5 ISA):
//  A (16x32 bf16): lanes 0-15 -> row=lane, elems 0..7 = K k0..k0+7, 8..15 = K k0+16..k0+23
//                  lanes16-31 -> row=lane-16, K k0+8..15 and k0+24..31
//  B (32x16 bf16): col = lane&15; lanes 0-15 hold K k0..k0+7, k0+16..k0+23; lanes16-31 K k0+8..15, k0+24..31
//  C (16x16 f32) : elem j, lane L -> row = j + 8*(L>=16), col = L&15
// ---------------------------------------------------------------------------
__device__ inline bf16x16 make_frag(bf16x8 lo, bf16x8 hi) {
  bf16x16 f;
#pragma unroll
  for (int i = 0; i < 8; ++i) { f[i] = lo[i]; f[i + 8] = hi[i]; }
  return f;
}

// A fragment through a lane-resident pointer (base already includes row and
// the +8 element offset for lanes 16..31); k0 steps the K dimension.
__device__ inline bf16x16 load_a_ptr(const __bf16* p, int k0) {
  bf16x8 lo = *(const bf16x8*)(p + k0);
  bf16x8 hi = *(const bf16x8*)(p + k0 + 16);
  return make_frag(lo, hi);
}

// B fragment through a lane-resident pointer (base includes column row and the
// +8 element offset for lanes 16..31).
__device__ inline bf16x16 load_b_ptr(const __bf16* p, int k0) {
  bf16x8 lo = *(const bf16x8*)(p + k0);
  bf16x8 hi = *(const bf16x8*)(p + k0 + 16);
  return make_frag(lo, hi);
}

#define WMMA_BF16(A_, B_, C_) \
  __builtin_amdgcn_wmma_f32_16x16x32_bf16(false, (A_), false, (B_), (short)0, (C_), false, false)

// ---------------------------------------------------------------------------
// Utility kernels
// ---------------------------------------------------------------------------
__global__ void f32_to_bf16_kernel(const float* __restrict__ in, __bf16* __restrict__ out, long n) {
  long stride = (long)gridDim.x * blockDim.x;
  for (long i = (long)blockIdx.x * blockDim.x + threadIdx.x; i < n; i += stride)
    out[i] = (__bf16)in[i];
}

__global__ void zero_f32_kernel(float* __restrict__ p, long n) {
  long stride = (long)gridDim.x * blockDim.x;
  for (long i = (long)blockIdx.x * blockDim.x + threadIdx.x; i < n; i += stride)
    p[i] = 0.f;
}

// ---------------------------------------------------------------------------
// Fused region kernel: per block = 32 tokens of one (view, batch).
// A tile (32x768) is staged once into LDS as bf16 (row stride 776 to spread
// banks); H = relu(X @ [imp_w1;clu_w1]^T + bias) via WMMA (N=1152 over 8
// waves), reduced to per-token importance + cluster logits, softmax,
// w = imp*clus, then atomic accumulation of num[b,r,d] / den[b,r].
// ---------------------------------------------------------------------------
#define ALDA 776  // padded LDS row stride for the 32x768 token tile

__global__ __launch_bounds__(256, 1) void region_kernel(
    const float* __restrict__ sat, const float* __restrict__ uav,
    const __bf16* __restrict__ iw1, const __bf16* __restrict__ cw1,
    const float* __restrict__ ib1, const float* __restrict__ cb1,
    const float* __restrict__ iw2, const float* __restrict__ ib2,
    const float* __restrict__ cw2, const float* __restrict__ cb2,
    float* __restrict__ num, float* __restrict__ den) {
  const int stile = blockIdx.x;   // 0..15  (32 tokens each, S=1024)
  const int b     = blockIdx.y;   // 0..31
  const int view  = blockIdx.z;   // 0..1
  const int tid   = threadIdx.x;
  const int lane  = tid & 31;
  const int wave  = tid >> 5;     // 0..7

  __shared__ __bf16 s_a[32 * ALDA];    // ~48.5 KB staged token tile (bf16)
  __shared__ float  s_iw2[384];
  __shared__ float  s_cw2[8 * 768];
  __shared__ float  s_bias[1152];
  __shared__ float  s_red[32 * 9];     // per token: [imp, logits 0..7]
  __shared__ float  s_w[32 * 8];

  const float* X = (view ? uav : sat) + ((long)b * 1024 + stile * 32) * 768;

  // stage A tile (fp32 -> bf16) once; every wave then reads it from LDS
  for (int i = tid; i < 32 * 768; i += 256) {
    int rr = i >> 9;          // i / 512? no: 768 per row -> use div
    rr = i / 768;
    int cc = i - rr * 768;
    s_a[rr * ALDA + cc] = (__bf16)X[(long)rr * 768 + cc];
  }
  for (int i = tid; i < 384; i += 256) s_iw2[i] = iw2[i];
  for (int i = tid; i < 8 * 768; i += 256) s_cw2[i] = cw2[i];
  for (int i = tid; i < 1152; i += 256) s_bias[i] = (i < 384) ? ib1[i] : cb1[i - 384];
  for (int i = tid; i < 32 * 9; i += 256) s_red[i] = 0.f;
  __syncthreads();

  f32x8 acc[2][9];
#pragma unroll
  for (int mt = 0; mt < 2; ++mt)
#pragma unroll
    for (int nt = 0; nt < 9; ++nt)
#pragma unroll
      for (int e = 0; e < 8; ++e) acc[mt][nt][e] = 0.f;

  // loop-invariant lane pointers (no per-iteration selects -> no WMMA hazards)
  const int n0 = wave * 144;  // 9 N-tiles of 16 per wave, N total = 1152
  const int kaoff = (lane & 16) ? 8 : 0;
  const __bf16* ap0 = &s_a[(lane & 15) * ALDA + kaoff];
  const __bf16* ap1 = &s_a[(16 + (lane & 15)) * ALDA + kaoff];
  const __bf16* bp[9];
#pragma unroll
  for (int nt = 0; nt < 9; ++nt) {
    int gn = n0 + nt * 16;
    const __bf16* wbase = (gn < 384) ? (iw1 + (long)gn * 768)
                                     : (cw1 + (long)(gn - 384) * 768);
    bp[nt] = wbase + (long)(lane & 15) * 768 + ((lane & 16) ? 8 : 0);
  }

  for (int k0 = 0; k0 < 768; k0 += 32) {
    bf16x16 a0 = load_a_ptr(ap0, k0);
    bf16x16 a1 = load_a_ptr(ap1, k0);
#pragma unroll
    for (int nt = 0; nt < 9; ++nt) {
      bf16x16 bf = load_b_ptr(bp[nt], k0);
      acc[0][nt] = WMMA_BF16(a0, bf, acc[0][nt]);
      acc[1][nt] = WMMA_BF16(a1, bf, acc[1][nt]);
    }
  }

  // bias+relu, dot with second-layer weights, reduce over this wave's N slice.
  const int col = lane & 15;
  const int hi8 = (lane & 16) ? 8 : 0;
#pragma unroll
  for (int mt = 0; mt < 2; ++mt) {
#pragma unroll
    for (int j = 0; j < 8; ++j) {
      int tok = mt * 16 + j + hi8;
      float impAcc = 0.f;
      float logAcc[8] = {0.f, 0.f, 0.f, 0.f, 0.f, 0.f, 0.f, 0.f};
#pragma unroll
      for (int nt = 0; nt < 9; ++nt) {
        int gn = n0 + nt * 16 + col;
        float v = fmaxf(acc[mt][nt][j] + s_bias[gn], 0.f);
        if (gn < 384) {
          impAcc += v * s_iw2[gn];
        } else {
          int nn = gn - 384;
#pragma unroll
          for (int r = 0; r < 8; ++r) logAcc[r] += v * s_cw2[r * 768 + nn];
        }
      }
      // butterfly reduce across the 16 columns (stays within lane halves)
#pragma unroll
      for (int m = 1; m < 16; m <<= 1) {
        impAcc += __shfl_xor(impAcc, m, 32);
#pragma unroll
        for (int r = 0; r < 8; ++r) logAcc[r] += __shfl_xor(logAcc[r], m, 32);
      }
      if (col == 0) {
        atomicAdd(&s_red[tok * 9 + 0], impAcc);
#pragma unroll
        for (int r = 0; r < 8; ++r) atomicAdd(&s_red[tok * 9 + 1 + r], logAcc[r]);
      }
    }
  }
  __syncthreads();

  // per-token softmax + weighted importance
  if (tid < 32) {
    int tok = tid;
    float imp = s_red[tok * 9] + ib2[0];
    float l[8];
    float mx = -1e30f;
#pragma unroll
    for (int r = 0; r < 8; ++r) { l[r] = s_red[tok * 9 + 1 + r] + cb2[r]; mx = fmaxf(mx, l[r]); }
    float sum = 0.f;
#pragma unroll
    for (int r = 0; r < 8; ++r) { l[r] = __expf(l[r] - mx); sum += l[r]; }
    float scale = imp / sum;
#pragma unroll
    for (int r = 0; r < 8; ++r) s_w[tok * 8 + r] = l[r] * scale;
  }
  __syncthreads();

  const long base = ((long)view * 32 + b) * 8;
  if (tid < 8) {
    float dsum = 0.f;
    for (int tok = 0; tok < 32; ++tok) dsum += s_w[tok * 8 + tid];
    atomicAdd(&den[base + tid], dsum);
  }

  // num[b,r,d] accumulation: thread owns d = tid, tid+256, tid+512 (reads the
  // LDS-staged tile instead of re-touching HBM)
  float p[8][3];
#pragma unroll
  for (int r = 0; r < 8; ++r)
#pragma unroll
    for (int c = 0; c < 3; ++c) p[r][c] = 0.f;
  for (int tok = 0; tok < 32; ++tok) {
    float f0 = (float)s_a[tok * ALDA + tid];
    float f1 = (float)s_a[tok * ALDA + tid + 256];
    float f2 = (float)s_a[tok * ALDA + tid + 512];
#pragma unroll
    for (int r = 0; r < 8; ++r) {
      float wr = s_w[tok * 8 + r];
      p[r][0] += wr * f0; p[r][1] += wr * f1; p[r][2] += wr * f2;
    }
  }
#pragma unroll
  for (int r = 0; r < 8; ++r) {
    float* np = num + (base + r) * 768;
    atomicAdd(&np[tid],       p[r][0]);
    atomicAdd(&np[tid + 256], p[r][1]);
    atomicAdd(&np[tid + 512], p[r][2]);
  }
}

// ---------------------------------------------------------------------------
// Generic WMMA GEMM: C(M,N) = act(A(M,K)_bf16 @ W(N,K)_bf16^T + bias)
// Block = 8 waves; tile M = MT*16 rows, N = 8*NT*16 cols.
// ACT: 0 = none, 1 = relu.  OBF: 0 = fp32 out, 1 = bf16 out.
// ---------------------------------------------------------------------------
template <int MT, int NT, int ACT, int OBF>
__global__ __launch_bounds__(256, 1) void gemm_bf16_kernel(
    const __bf16* __restrict__ A, const __bf16* __restrict__ W,
    const float* __restrict__ bias, void* __restrict__ Cout, int N, int K) {
  const int lane = threadIdx.x & 31;
  const int wave = threadIdx.x >> 5;
  const int m0 = blockIdx.x * (MT * 16);
  const int n0 = blockIdx.y * (8 * NT * 16) + wave * (NT * 16);

  f32x8 acc[MT][NT];
#pragma unroll
  for (int mt = 0; mt < MT; ++mt)
#pragma unroll
    for (int nt = 0; nt < NT; ++nt)
#pragma unroll
      for (int e = 0; e < 8; ++e) acc[mt][nt][e] = 0.f;

  // loop-invariant lane pointers
  const __bf16* ap[MT];
#pragma unroll
  for (int mt = 0; mt < MT; ++mt)
    ap[mt] = A + (long)(m0 + mt * 16 + (lane & 15)) * K + ((lane & 16) ? 8 : 0);
  const __bf16* bp[NT];
#pragma unroll
  for (int nt = 0; nt < NT; ++nt)
    bp[nt] = W + (long)(n0 + nt * 16 + (lane & 15)) * K + ((lane & 16) ? 8 : 0);

  for (int k0 = 0; k0 < K; k0 += 32) {
    bf16x16 a[MT];
#pragma unroll
    for (int mt = 0; mt < MT; ++mt) a[mt] = load_a_ptr(ap[mt], k0);
#pragma unroll
    for (int nt = 0; nt < NT; ++nt) {
      bf16x16 bf = load_b_ptr(bp[nt], k0);
#pragma unroll
      for (int mt = 0; mt < MT; ++mt) acc[mt][nt] = WMMA_BF16(a[mt], bf, acc[mt][nt]);
    }
  }

  const int col = lane & 15;
  const int hi8 = (lane & 16) ? 8 : 0;
#pragma unroll
  for (int mt = 0; mt < MT; ++mt)
#pragma unroll
    for (int nt = 0; nt < NT; ++nt)
#pragma unroll
      for (int j = 0; j < 8; ++j) {
        int row = m0 + mt * 16 + j + hi8;
        int c   = n0 + nt * 16 + col;
        float v = acc[mt][nt][j] + (bias ? bias[c] : 0.f);
        if (ACT == 1) v = fmaxf(v, 0.f);
        if (OBF) ((__bf16*)Cout)[(long)row * N + c] = (__bf16)v;
        else     ((float*)Cout)[(long)row * N + c]  = v;
      }
}

// ---------------------------------------------------------------------------
// regions[v][b][r][:] = num / (den + 1e-8)
// ---------------------------------------------------------------------------
__global__ void finalize_regions_kernel(const float* __restrict__ num,
                                        const float* __restrict__ den,
                                        float* __restrict__ sat_reg,
                                        float* __restrict__ uav_reg) {
  int blk = blockIdx.x;           // v*256 + b*8 + r  (512 total)
  int v = blk >> 8;
  int br = blk & 255;
  float inv = 1.f / (den[blk] + 1e-8f);
  float* out = (v ? uav_reg : sat_reg) + (long)br * 768;
  const float* in = num + (long)blk * 768;
  for (int d = threadIdx.x; d < 768; d += 256) out[d] = in[d] * inv;
}

// pair[b,i,j,:] = [sat_r[b,i,:], uav_r[b,j,:]]  (bf16, 2048 x 1536)
__global__ void pair_prep_kernel(const float* __restrict__ sat_reg,
                                 const float* __restrict__ uav_reg,
                                 __bf16* __restrict__ out) {
  const long total = (long)2048 * 1536;
  long stride = (long)gridDim.x * blockDim.x;
  for (long idx = (long)blockIdx.x * blockDim.x + threadIdx.x; idx < total; idx += stride) {
    int c = (int)(idx % 1536);
    long row = idx / 1536;
    int b = (int)(row >> 6);
    int i = (int)((row >> 3) & 7);
    int j = (int)(row & 7);
    float v = (c < 768) ? sat_reg[((long)b * 8 + i) * 768 + c]
                        : uav_reg[((long)b * 8 + j) * 768 + (c - 768)];
    out[idx] = (__bf16)v;
  }
}

// sim[i,j] = mean_b sigmoid(h2[b,i,j,:] @ sim_w3 + b3)
__global__ void sim_reduce_kernel(const float* __restrict__ h2,
                                  const float* __restrict__ w3,
                                  const float* __restrict__ b3,
                                  float* __restrict__ sim) {
  __shared__ float red[256];
  const int t = threadIdx.x;
  float s = 0.f;
  for (int b = 0; b < 32; ++b) {
    const float* x = h2 + ((long)b * 64 + blockIdx.x) * 384;
    float p = x[t] * w3[t];
    if (t < 128) p += x[256 + t] * w3[256 + t];
    red[t] = p; __syncthreads();
    for (int k = 128; k > 0; k >>= 1) { if (t < k) red[t] += red[t + k]; __syncthreads(); }
    if (t == 0) s += 1.f / (1.f + __expf(-(red[0] + b3[0])));
    __syncthreads();
  }
  if (t == 0) sim[blockIdx.x] = s * (1.f / 32.f);
}

// greedy top-1 matching without replacement (R = 8)
__global__ void greedy_kernel(const float* __restrict__ sim, int* __restrict__ perm) {
  if (threadIdx.x == 0) {
    unsigned used = 0;
    for (int i = 0; i < 8; ++i) {
      float best = -2.f; int bj = 0;
      for (int j = 0; j < 8; ++j) {
        float v = ((used >> j) & 1u) ? -1.f : sim[i * 8 + j];
        if (v > best) { best = v; bj = j; }
      }
      used |= 1u << bj;
      perm[i] = bj;
    }
  }
}

// asum[b*8+r,:] = sat_r[b,r,:] + uav_r[b,perm[r],:]  (bf16)
__global__ void asum_prep_kernel(const float* __restrict__ sat_reg,
                                 const float* __restrict__ uav_reg,
                                 const int* __restrict__ perm,
                                 __bf16* __restrict__ out) {
  int idx = blockIdx.x * 256 + threadIdx.x;   // 256*768 total, grid = 768
  int d = idx % 768;
  int br = idx / 768;
  int b = br >> 3, r = br & 7;
  float v = sat_reg[idx] + uav_reg[((long)b * 8 + perm[r]) * 768 + d];
  out[idx] = (__bf16)v;
}

// bias_eff[n] = 2*(wo[n,:] . bv) + 2*bo[n]
__global__ void bias_eff_kernel(const float* __restrict__ wo, const float* __restrict__ bv,
                                const float* __restrict__ bo, float* __restrict__ out) {
  int n = blockIdx.x * 256 + threadIdx.x;
  if (n >= 768) return;
  float s = 0.f;
  for (int d = 0; d < 768; ++d) s += wo[(long)n * 768 + d] * bv[d];
  out[n] = 2.f * s + 2.f * bo[n];
}

// row-wise LayerNorm; OBF -> bf16 output, RELU -> relu after affine
template <int OBF, int RELU>
__global__ void ln_kernel(const float* __restrict__ X, const float* __restrict__ g,
                          const float* __restrict__ bta, void* __restrict__ out, int cols) {
  const int row = blockIdx.x, t = threadIdx.x;
  const float* x = X + (long)row * cols;
  __shared__ float red[256];
  __shared__ float s_mean, s_rstd;
  float s = 0.f;
  for (int c = t; c < cols; c += 256) s += x[c];
  red[t] = s; __syncthreads();
  for (int k = 128; k > 0; k >>= 1) { if (t < k) red[t] += red[t + k]; __syncthreads(); }
  if (t == 0) s_mean = red[0] / (float)cols;
  __syncthreads();
  float m = s_mean;
  s = 0.f;
  for (int c = t; c < cols; c += 256) { float d = x[c] - m; s += d * d; }
  red[t] = s; __syncthreads();
  for (int k = 128; k > 0; k >>= 1) { if (t < k) red[t] += red[t + k]; __syncthreads(); }
  if (t == 0) s_rstd = rsqrtf(red[0] / (float)cols + 1e-5f);
  __syncthreads();
  float rs = s_rstd;
  for (int c = t; c < cols; c += 256) {
    float v = (x[c] - m) * rs * g[c] + bta[c];
    if (RELU) v = fmaxf(v, 0.f);
    if (OBF) ((__bf16*)out)[(long)row * cols + c] = (__bf16)v;
    else     ((float*)out)[(long)row * cols + c]  = v;
  }
}

// ---------------------------------------------------------------------------
// Host orchestration
// ---------------------------------------------------------------------------
extern "C" void kernel_launch(void* const* d_in, const int* in_sizes, int n_in,
                              void* d_out, int out_size, void* d_ws, size_t ws_size,
                              hipStream_t stream) {
  (void)in_sizes; (void)n_in; (void)out_size; (void)ws_size;
  const float* sat    = (const float*)d_in[0];
  const float* uav    = (const float*)d_in[1];
  const float* imp_w1 = (const float*)d_in[2];
  const float* imp_b1 = (const float*)d_in[3];
  const float* imp_w2 = (const float*)d_in[4];
  const float* imp_b2 = (const float*)d_in[5];
  const float* clu_w1 = (const float*)d_in[6];
  const float* clu_b1 = (const float*)d_in[7];
  const float* clu_w2 = (const float*)d_in[8];
  const float* clu_b2 = (const float*)d_in[9];
  const float* sim_w1 = (const float*)d_in[10];
  const float* sim_b1 = (const float*)d_in[11];
  const float* sim_w2 = (const float*)d_in[12];
  const float* sim_b2 = (const float*)d_in[13];
  const float* sim_w3 = (const float*)d_in[14];
  const float* sim_b3 = (const float*)d_in[15];
  // d_in[16..19] = wq,bq,wk,bk: mathematically inert (kv len == 1), unused.
  const float* wv     = (const float*)d_in[20];
  const float* bv     = (const float*)d_in[21];
  const float* wo     = (const float*)d_in[22];
  const float* bo     = (const float*)d_in[23];
  const float* ln_g   = (const float*)d_in[24];
  const float* ln_b   = (const float*)d_in[25];
  const float* fus_w  = (const float*)d_in[26];
  const float* fus_b  = (const float*)d_in[27];
  const float* fus_g  = (const float*)d_in[28];
  const float* fus_bs = (const float*)d_in[29];

  char* wsb = (char*)d_ws;
  size_t off = 0;
  auto carve = [&](size_t bytes) -> void* {
    void* p = wsb + off;
    off += (bytes + 255) & ~(size_t)255;
    return p;
  };

  __bf16* iw1_bf  = (__bf16*)carve((size_t)384 * 768 * 2);
  __bf16* cw1_bf  = (__bf16*)carve((size_t)768 * 768 * 2);
  __bf16* sw1_bf  = (__bf16*)carve((size_t)768 * 1536 * 2);
  __bf16* sw2_bf  = (__bf16*)carve((size_t)384 * 768 * 2);
  __bf16* wv_bf   = (__bf16*)carve((size_t)768 * 768 * 2);
  __bf16* wo_bf   = (__bf16*)carve((size_t)768 * 768 * 2);
  __bf16* fusw_bf = (__bf16*)carve((size_t)768 * 6144 * 2);
  float*  num     = (float*)carve((size_t)2 * 32 * 8 * 768 * 4);
  float*  den     = (float*)carve((size_t)2 * 32 * 8 * 4);
  float*  sat_reg = (float*)carve((size_t)32 * 8 * 768 * 4);
  float*  uav_reg = (float*)carve((size_t)32 * 8 * 768 * 4);
  __bf16* pair_bf = (__bf16*)carve((size_t)2048 * 1536 * 2);
  __bf16* h1_bf   = (__bf16*)carve((size_t)2048 * 768 * 2);
  float*  h2      = (float*)carve((size_t)2048 * 384 * 4);
  float*  sim     = (float*)carve((size_t)64 * 4);
  int*    perm    = (int*)carve((size_t)8 * 4);
  __bf16* asum_bf = (__bf16*)carve((size_t)256 * 768 * 2);
  __bf16* t_bf    = (__bf16*)carve((size_t)256 * 768 * 2);
  float*  y_f32   = (float*)carve((size_t)256 * 768 * 4);
  __bf16* pool_bf = (__bf16*)carve((size_t)256 * 768 * 2);
  float*  biaseff = (float*)carve((size_t)768 * 4);
  float*  fuseout = (float*)carve((size_t)32 * 768 * 4);

  auto cvt = [&](const float* src, __bf16* dst, long n) {
    int grid = (int)((n + 255) / 256);
    if (grid > 2048) grid = 2048;
    f32_to_bf16_kernel<<<grid, 256, 0, stream>>>(src, dst, n);
  };
  // 1) weights fp32 -> bf16 (stay hot in 192MB L2 across the whole pipeline)
  cvt(imp_w1, iw1_bf, (long)384 * 768);
  cvt(clu_w1, cw1_bf, (long)768 * 768);
  cvt(sim_w1, sw1_bf, (long)768 * 1536);
  cvt(sim_w2, sw2_bf, (long)384 * 768);
  cvt(wv, wv_bf, (long)768 * 768);
  cvt(wo, wo_bf, (long)768 * 768);
  cvt(fus_w, fusw_bf, (long)768 * 6144);

  // 2) zero accumulators
  zero_f32_kernel<<<1536, 256, 0, stream>>>(num, (long)2 * 32 * 8 * 768);
  zero_f32_kernel<<<2, 256, 0, stream>>>(den, 512);
  zero_f32_kernel<<<1, 256, 0, stream>>>(sim, 64);

  // 3) dominant fused region kernel (both views): WMMA MLP + softmax + region accum
  region_kernel<<<dim3(16, 32, 2), 256, 0, stream>>>(
      sat, uav, iw1_bf, cw1_bf, imp_b1, clu_b1, imp_w2, imp_b2, clu_w2, clu_b2, num, den);

  // 4) regions = num / (den + 1e-8)
  finalize_regions_kernel<<<512, 256, 0, stream>>>(num, den, sat_reg, uav_reg);

  // 5) pairwise similarity MLP
  pair_prep_kernel<<<2048, 256, 0, stream>>>(sat_reg, uav_reg, pair_bf);
  gemm_bf16_kernel<2, 6, 1, 1><<<dim3(64, 1), 256, 0, stream>>>(
      pair_bf, sw1_bf, sim_b1, h1_bf, 768, 1536);          // h1 = relu(pair@W1^T+b1), bf16
  gemm_bf16_kernel<2, 3, 1, 0><<<dim3(64, 1), 256, 0, stream>>>(
      h1_bf, sw2_bf, sim_b2, h2, 384, 768);                // h2 = relu(h1@W2^T+b2), fp32
  sim_reduce_kernel<<<64, 256, 0, stream>>>(h2, sim_w3, sim_b3, sim);

  // 6) greedy matching -> perm
  greedy_kernel<<<1, 32, 0, stream>>>(sim, perm);

  // 7) attention pooling (len-1 kv => linear): y = ((sat_r+uav_al)@Wv^T)@Wo^T + 2(Wo.bv)+2bo
  asum_prep_kernel<<<768, 256, 0, stream>>>(sat_reg, uav_reg, perm, asum_bf);
  bias_eff_kernel<<<3, 256, 0, stream>>>(wo, bv, bo, biaseff);
  gemm_bf16_kernel<2, 6, 0, 1><<<dim3(8, 1), 256, 0, stream>>>(
      asum_bf, wv_bf, (const float*)nullptr, t_bf, 768, 768);
  gemm_bf16_kernel<2, 6, 0, 0><<<dim3(8, 1), 256, 0, stream>>>(
      t_bf, wo_bf, biaseff, y_f32, 768, 768);
  ln_kernel<1, 0><<<256, 256, 0, stream>>>(y_f32, ln_g, ln_b, pool_bf, 768);  // pooled LN -> bf16

  // 8) region fusion: (32 x 6144) @ fus_w^T + fus_b, then LN + relu -> d_out
  gemm_bf16_kernel<2, 1, 0, 0><<<dim3(1, 6), 256, 0, stream>>>(
      pool_bf, fusw_bf, fus_b, fuseout, 768, 6144);
  ln_kernel<0, 1><<<32, 256, 0, stream>>>(fuseout, fus_g, fus_bs, (float*)d_out, 768);
}
